// EdgeNetwork_13116830122450
// MI455X (gfx1250) — matrix operand order repaired and checked by reading, not compiled, not run on hardware
//
#include <hip/hip_runtime.h>
#include <hip/hip_bf16.h>

// EdgeNetwork fused kernel for MI455X (gfx1250, wave32, WMMA).
//
// Math:  t[e,i] = sum_{k,j} bond[e,k] * K[k, i*32+j] * nbr[e,j]  + bias-term
// Factored as one GEMM through the per-edge outer product:
//   O_ext[e, ks*32+j] = bond'[e,ks] * nbr[e,j]    (bond'[e,16] == 1 -> bias row)
//   K'_ext[ks*32+j, i] = (ks<16) ? kernel[ks, i*32+j] : bias[i*32+j]
//   T = O_ext (E x 544) @ K'_ext (544 x 32),  K = 544 = 17 * 32
// -> 17 chained V_WMMA_F32_16X16X32_F16 per 16-edge tile per 16-col ntile.
//
// Round-2 change: ALL 17 A-fragments are materialized in registers before the
// WMMA loop, so the steady-state loop is ds_load_b128 + v_wmma only — no VALU
// writes into live WMMA source registers, eliminating the 4x v_nop
// WMMA->VALU WAR hazard bubbles the previous version exhibited.

typedef __attribute__((ext_vector_type(16))) _Float16 v16h;
typedef __attribute__((ext_vector_type(8)))  float    v8f;

#define N_ATOMS   100000
#define N_EDGES   400000
#define ATOM_DIM  32
#define BOND_DIM  16
#define KC        544                      // 17 k-steps * 32
#define KCP       544                      // LDS pitch (halves) per output column
#define WAVES_PER_BLOCK 8
#define THREADS   256
#define TILES_PER_WAVE  5
#define NUM_TILES (N_EDGES / 16)           // 25000
#define NUM_BLOCKS (NUM_TILES / (TILES_PER_WAVE * WAVES_PER_BLOCK)) // 625

__device__ __forceinline__ v16h splat16(_Float16 x) {
    v16h r = {x,x,x,x,x,x,x,x,x,x,x,x,x,x,x,x};
    return r;
}

__global__ __launch_bounds__(THREADS)
void edge_network_wmma(const float* __restrict__ atom,
                       const float* __restrict__ bond,
                       const int*   __restrict__ pair32,  // int64 pairs viewed as i32
                       const float* __restrict__ kern,
                       const float* __restrict__ bias,
                       float*       __restrict__ out)
{
    // Fused K'_ext in LDS, f16, column-major over output column i:
    //   KcLds[i*KCP + kc],  kc = ks*32 + j
    extern __shared__ _Float16 KcLds[] __attribute__((aligned(32)));

    const int tid = threadIdx.x;

    // ---- Stage K'_ext (kernel reshaped + bias block), f32 -> f16 ----
    for (int t = tid; t < ATOM_DIM * KC; t += THREADS) {
        const int kc = t >> 5;         // 0..543
        const int i  = t & 31;         // output column
        float v;
        if (kc < 512) {
            // kernel[k][i*32 + j], k = kc>>5, j = kc&31
            v = kern[((kc >> 5) << 10) + (i << 5) + (kc & 31)];
        } else {
            v = bias[(i << 5) + (kc & 31)];
        }
        KcLds[i * KCP + kc] = (_Float16)v;
    }
    __syncthreads();

    const int lane = tid & 31;
    const int half = lane >> 4;        // which 16-lane half of the wave32
    const int m    = lane & 15;        // row (edge-in-tile) for A, col for B/C/D
    const int wave = tid >> 5;

    for (int t = 0; t < TILES_PER_WAVE; ++t) {
        const int tile = (blockIdx.x * WAVES_PER_BLOCK + wave) * TILES_PER_WAVE + t;
        if (tile >= NUM_TILES) break;
        const int e0 = tile << 4;      // first edge of this 16-edge tile
        const int em = e0 + m;

        // Scatter (dst) indices for this lane's 8 output rows — load early so
        // the epilogue atomics are not load-latency bound.
        int dstv[8];
        #pragma unroll
        for (int v = 0; v < 8; ++v) {
            dstv[v] = pair32[(e0 + v + (half << 3)) << 2];  // low dword of pair[e][0]
        }

        // Neighbor row for edge em (gathered through L2; atoms fit in 192MB L2).
        // A uses contiguous k here (same as B): half0 holds K={0..15}, half1 K={16..31}.
        const int src = pair32[(em << 2) + 2];           // low dword of pair[em][1]
        const float4* rowv = (const float4*)(atom + ((size_t)src << 5));
        const float4 x0 = rowv[4 * half + 0];            // j = 16*half + 0..3
        const float4 x1 = rowv[4 * half + 1];            // j = 16*half + 4..7
        const float4 x2 = rowv[4 * half + 2];            // j = 16*half + 8..11
        const float4 x3 = rowv[4 * half + 3];            // j = 16*half + 12..15
        v16h nf;
        nf[0]  = (_Float16)x0.x;  nf[1]  = (_Float16)x0.y;
        nf[2]  = (_Float16)x0.z;  nf[3]  = (_Float16)x0.w;
        nf[4]  = (_Float16)x1.x;  nf[5]  = (_Float16)x1.y;
        nf[6]  = (_Float16)x1.z;  nf[7]  = (_Float16)x1.w;
        nf[8]  = (_Float16)x2.x;  nf[9]  = (_Float16)x2.y;
        nf[10] = (_Float16)x2.z;  nf[11] = (_Float16)x2.w;
        nf[12] = (_Float16)x3.x;  nf[13] = (_Float16)x3.y;
        nf[14] = (_Float16)x3.z;  nf[15] = (_Float16)x3.w;

        // Bond row (16 floats) for edge em, converted to f16 once.
        const float4* bv = (const float4*)(bond + ((size_t)em << 4));
        const float4 b0 = bv[0], b1 = bv[1], b2 = bv[2], b3 = bv[3];
        const float brf[16] = { b0.x, b0.y, b0.z, b0.w,  b1.x, b1.y, b1.z, b1.w,
                                b2.x, b2.y, b2.z, b2.w,  b3.x, b3.y, b3.z, b3.w };
        _Float16 bh[16];
        #pragma unroll
        for (int ks = 0; ks < 16; ++ks) bh[ks] = (_Float16)brf[ks];

        // ---- Materialize ALL 17 A-fragments up front (17*8 = 136 VGPRs).
        // Steady-state WMMA loop then has no VALU writes into live WMMA
        // sources -> no WMMA->VALU WAR hazard NOPs.
        v16h afrag[17];
        #pragma unroll
        for (int ks = 0; ks < 16; ++ks) afrag[ks] = nf * splat16(bh[ks]);
        afrag[16] = nf;   // implicit "1" row -> bias contribution

        v8f acc0 = {};    // output columns i = 0..15
        v8f acc1 = {};    // output columns i = 16..31

        const _Float16* colA = &KcLds[(size_t)m        * KCP + (half << 4)];
        const _Float16* colB = &KcLds[(size_t)(16 + m) * KCP + (half << 4)];

        #pragma unroll
        for (int ks = 0; ks < 17; ++ks) {
            // B fragment: 16 consecutive K values for this lane's column,
            // half0 -> K = ks*32 + 0..15, half1 -> K = ks*32 + 16..31
            const v16h bA = *(const v16h*)(colA + (ks << 5));
            const v16h bB = *(const v16h*)(colB + (ks << 5));
            acc0 = __builtin_amdgcn_wmma_f32_16x16x32_f16(
                       false, afrag[ks], false, bA, (short)0, acc0, false, false);
            acc1 = __builtin_amdgcn_wmma_f32_16x16x32_f16(
                       false, afrag[ks], false, bB, (short)0, acc1, false, false);
        }

        // Scatter-add. C/D layout: VGPR v holds edge (v + 8*half), column m.
        #pragma unroll
        for (int v = 0; v < 8; ++v) {
            float* o = out + ((size_t)dstv[v] << 5);
            atomicAdd(o + m,      acc0[v]);
            atomicAdd(o + 16 + m, acc1[v]);
        }
    }
}

extern "C" void kernel_launch(void* const* d_in, const int* in_sizes, int n_in,
                              void* d_out, int out_size, void* d_ws, size_t ws_size,
                              hipStream_t stream) {
    const float* atom   = (const float*)d_in[0];   // (100000, 32) f32
    const float* bond   = (const float*)d_in[1];   // (400000, 16) f32
    const int*   pair32 = (const int*)  d_in[2];   // (400000, 2) int64 viewed as i32
    const float* kern   = (const float*)d_in[3];   // (16, 1024) f32
    const float* bias   = (const float*)d_in[4];   // (1024,) f32
    float* out = (float*)d_out;                    // (100000, 32) f32

    hipMemsetAsync(d_out, 0, (size_t)out_size * sizeof(float), stream);

    const size_t shmem = (size_t)ATOM_DIM * KC * sizeof(_Float16);  // 34816 B
    edge_network_wmma<<<NUM_BLOCKS, THREADS, shmem, stream>>>(
        atom, bond, pair32, kern, bias, out);
}
